// SWTPrecomputedMarketStateEncoder_44822278701157
// MI455X (gfx1250) — matrix operand order repaired and checked by reading, not compiled, not run on hardware
//
#include <hip/hip_runtime.h>

typedef __attribute__((ext_vector_type(16))) __bf16 v16bf;
typedef __attribute__((ext_vector_type(8)))  float  v8f;
typedef __attribute__((ext_vector_type(4)))  float  v4f;

#define LDS_STRIDE 140   // 137 cols + pad, avoids LDS bank conflicts
#define WAVES_PER_BLOCK 4

// Swizzled A-operand weight table layout in d_ws (bf16):
//   L1: [t(4)][lane(32)][e(16)]            at offset 0      (2048 elems, K-pad zeros baked in)
//   L2: [t(8)][b(2)][lane(32)][e(16)]      at offset 2048   (8192 elems)
//   L3: [t(8)][b(4)][lane(32)][e(16)]      at offset 10240  (16384 elems)
// Each lane's 16 elements are its exact WMMA A-operand contents (chaining
// permutation w(t,m)=32(t>>1)+16(m>>3)+8(t&1)+(m&7) folded in for L1/L2),
// so the hot kernel does one contiguous 32B load per A operand.
__global__ __launch_bounds__(256) void wswizzle_kernel(
    const float* __restrict__ W1, const float* __restrict__ W2,
    const float* __restrict__ W3, __bf16* __restrict__ wh)
{
  int i = blockIdx.x * 256 + threadIdx.x;
  if (i >= 26624) return;
  float v = 0.0f;
  if (i < 2048) {                       // layer 1 (K=16, padded to 32)
    int j = i;
    int e = j & 15, lane = (j >> 4) & 31, t = j >> 9;
    int l = lane & 15, g = lane >> 4;
    int wrow = 32 * (t >> 1) + 16 * (l >> 3) + 8 * (t & 1) + (l & 7);
    if (g == 0) v = W1[wrow * 16 + e];      // g==1 -> zero pad
  } else if (i < 10240) {               // layer 2
    int j = i - 2048;
    int e = j & 15, lane = (j >> 4) & 31, b = (j >> 9) & 1, t = j >> 10;
    int l = lane & 15, g = lane >> 4;
    int wrow = 32 * (t >> 1) + 16 * (l >> 3) + 8 * (t & 1) + (l & 7);
    int kl = 16 * g + e;
    v = W2[wrow * 64 + 32 * b + kl];
  } else {                              // layer 3 (natural row order)
    int j = i - 10240;
    int e = j & 15, lane = (j >> 4) & 31, b = (j >> 9) & 3, t = j >> 11;
    int g = lane >> 4;
    int wrow = 16 * t + (lane & 15);
    int kl = 16 * g + e;
    v = W3[wrow * 128 + 32 * b + kl];
  }
  wh[i] = (__bf16)v;
}

// Each wave handles 16 batch rows through the full MLP using the transposed
// formulation  g = W @ h^T  so WMMA D-registers chain directly into the next
// layer's B-operand after bf16 packing (no LDS / shuffles between layers).
__global__ __launch_bounds__(128) void encoder_kernel(
    const float* __restrict__ wst, const float* __restrict__ pos,
    const float* __restrict__ b1,  const float* __restrict__ b2,
    const float* __restrict__ b3,  const int* __restrict__ idx,
    const __bf16* __restrict__ wh, float* __restrict__ out, int nrows)
{
  __shared__ float stage[WAVES_PER_BLOCK * 16 * LDS_STRIDE];
  const int lane = threadIdx.x & 31;
  const int wave = threadIdx.x >> 5;
  const int l = lane & 15;        // row (A) / batch column (B,D)
  const int g = lane >> 4;        // lane half
  const int rowBase = (blockIdx.x * WAVES_PER_BLOCK + wave) * 16;
  float* sl = stage + wave * (16 * LDS_STRIDE);

  // per-lane base into the swizzled A-operand table
  const __bf16* whl = wh + (lane << 4);

  // ---- gather: build B-operand X^T (16 real K, padded to 32) ----
  // B layout: lanes 0-15 hold k=0..15 (all 16 features), lanes 16-31 zero pad
  v16bf B0;
  #pragma unroll
  for (int e = 0; e < 16; ++e) B0[e] = (__bf16)0.0f;
  {
    int rr = rowBase + l; if (rr >= nrows) rr = nrows - 1;
    int trow = idx[rr];
    if (g == 0) {
      const v4f* src = (const v4f*)(wst + (size_t)trow * 16);
      v4f x0 = src[0], x1 = src[1], x2 = src[2], x3 = src[3];
      #pragma unroll
      for (int e = 0; e < 4; ++e) {
        B0[e]      = (__bf16)x0[e];
        B0[4 + e]  = (__bf16)x1[e];
        B0[8 + e]  = (__bf16)x2[e];
        B0[12 + e] = (__bf16)x3[e];
      }
    }
  }

  // ---- layer 1: h1 = relu(W1 @ x^T + b1), 64 features -> B1[2] operands ----
  v16bf B1[2];
  #pragma unroll
  for (int p = 0; p < 2; ++p) {
    float hh[2][8];
    #pragma unroll
    for (int q = 0; q < 2; ++q) {
      const int t = 2 * p + q;
      v16bf A = *(const v16bf*)(whl + t * (32 * 16));
      v8f acc = {0.f,0.f,0.f,0.f,0.f,0.f,0.f,0.f};
      acc = __builtin_amdgcn_wmma_f32_16x16x32_bf16(false, A, false, B0,
                                                    (short)0, acc, false, false);
      const float* bb = b1 + 32 * (t >> 1) + 16 * g + 8 * (t & 1);
      #pragma unroll
      for (int r = 0; r < 8; ++r) hh[q][r] = fmaxf(acc[r] + bb[r], 0.0f);
    }
    #pragma unroll
    for (int e = 0; e < 16; ++e) B1[p][e] = (__bf16)hh[e >> 3][e & 7];
  }

  // ---- layer 2: h2 = relu(W2 @ h1^T + b2), 128 features -> B2[4] operands ----
  v16bf B2[4];
  #pragma unroll
  for (int p = 0; p < 4; ++p) {
    float hh[2][8];
    #pragma unroll
    for (int q = 0; q < 2; ++q) {
      const int t = 2 * p + q;
      v8f acc = {0.f,0.f,0.f,0.f,0.f,0.f,0.f,0.f};
      #pragma unroll
      for (int b = 0; b < 2; ++b) {
        v16bf A = *(const v16bf*)(whl + 2048 + (t * 2 + b) * (32 * 16));
        acc = __builtin_amdgcn_wmma_f32_16x16x32_bf16(false, A, false, B1[b],
                                                      (short)0, acc, false, false);
      }
      const float* bb = b2 + 32 * (t >> 1) + 16 * g + 8 * (t & 1);
      #pragma unroll
      for (int r = 0; r < 8; ++r) hh[q][r] = fmaxf(acc[r] + bb[r], 0.0f);
    }
    #pragma unroll
    for (int e = 0; e < 16; ++e) B2[p][e] = (__bf16)hh[e >> 3][e & 7];
  }

  // ---- layer 3: out = W3 @ h2^T + b3, stage rows in LDS for coalesced store ----
  #pragma unroll
  for (int t = 0; t < 8; ++t) {
    v8f acc = {0.f,0.f,0.f,0.f,0.f,0.f,0.f,0.f};
    #pragma unroll
    for (int b = 0; b < 4; ++b) {
      v16bf A = *(const v16bf*)(whl + 10240 + (t * 4 + b) * (32 * 16));
      acc = __builtin_amdgcn_wmma_f32_16x16x32_bf16(false, A, false, B2[b],
                                                    (short)0, acc, false, false);
    }
    const float* bb = b3 + 16 * t + 8 * g;
    float* dst = sl + l * LDS_STRIDE + 16 * t + 8 * g;   // contiguous 8 floats
    #pragma unroll
    for (int r = 0; r < 8; ++r) dst[r] = acc[r] + bb[r];
  }

  // ---- position passthrough into staging cols 128..136 ----
  for (int j = lane; j < 16 * 9; j += 32) {
    int r = j / 9, c = j - r * 9;
    int rr = rowBase + r; if (rr >= nrows) rr = nrows - 1;
    sl[r * LDS_STRIDE + 128 + c] = pos[(size_t)rr * 9 + c];
  }

  // Per-wave private staging; LDS ops from one wave are processed in order,
  // so no block barrier is needed before reading back.
  for (int j = lane; j < 16 * 137; j += 32) {
    int r = j / 137, c = j - r * 137;
    int rr = rowBase + r;
    if (rr < nrows) out[(size_t)rr * 137 + c] = sl[r * LDS_STRIDE + c];
  }
}

extern "C" void kernel_launch(void* const* d_in, const int* in_sizes, int n_in,
                              void* d_out, int out_size, void* d_ws, size_t ws_size,
                              hipStream_t stream) {
  (void)n_in; (void)out_size; (void)ws_size;
  const float* wst = (const float*)d_in[0];
  const float* pos = (const float*)d_in[1];
  const float* W1  = (const float*)d_in[2];
  const float* b1  = (const float*)d_in[3];
  const float* W2  = (const float*)d_in[4];
  const float* b2  = (const float*)d_in[5];
  const float* W3  = (const float*)d_in[6];
  const float* b3  = (const float*)d_in[7];
  const int*   idx = (const int*)d_in[8];
  float* out = (float*)d_out;
  __bf16* wh = (__bf16*)d_ws;          // 26,624 bf16 = 53,248 B scratch

  const int nrows = in_sizes[8];

  wswizzle_kernel<<<dim3(104), dim3(256), 0, stream>>>(W1, W2, W3, wh);

  const int rowsPerBlock = WAVES_PER_BLOCK * 16;
  const int nblocks = (nrows + rowsPerBlock - 1) / rowsPerBlock;
  encoder_kernel<<<dim3(nblocks), dim3(32 * WAVES_PER_BLOCK), 0, stream>>>(
      wst, pos, b1, b2, b3, idx, wh, out, nrows);
}
